// NodeEarlyInteractionWithConsistency_65257733095600
// MI455X (gfx1250) — matrix operand order repaired and checked by reading, not compiled, not run on hardware
//
#include <hip/hip_runtime.h>

typedef __attribute__((ext_vector_type(16))) _Float16 v16h;
typedef __attribute__((ext_vector_type(8)))  float    v8f;

#define DEV __device__ __forceinline__

constexpr int kNodes = 65536;   // G2 * N
constexpr int kEdges = 131072;  // G2 * EPG
constexpr int kG     = 1024;    // graph pairs
constexpr float kInvTemp = 10.0f; // 1/TEMP

// ---------------------------------------------------------------------------
// WMMA helpers (CDNA5, wave32, V_WMMA_F32_16X16X32_F16)
// ---------------------------------------------------------------------------
DEV v8f vzero() {
  v8f z;
#pragma unroll
  for (int i = 0; i < 8; ++i) z[i] = 0.f;
  return z;
}

DEV v8f wmma16(v16h a, v16h b, v8f c) {
  return __builtin_amdgcn_wmma_f32_16x16x32_f16(
      false, a, false, b, (short)0, c, false, false);
}

// B fragment: weights pre-packed transposed as Wt[N][Kpad] in f16.
// Lane layout (32x16 f16 B): lane l -> col c=l&15, group g=l>>4,
// b[j] = B[kt*32 + (j>>3)*16 + g*8 + (j&7)][c], j=0..15.
DEV v16h loadB(const _Float16* __restrict__ Wt, int Kpad, int ct, int kt) {
  const int lane = threadIdx.x & 31;
  const int c = lane & 15, g = lane >> 4;
  const _Float16* p = Wt + (size_t)((ct << 4) | c) * Kpad + kt * 32 + g * 8;
  v16h b;
#pragma unroll
  for (int i = 0; i < 16; ++i) b[i] = p[(i >> 3) * 16 + (i & 7)];
  return b;
}

// A fragment from per-wave LDS hidden tile [16][ldh] f16.
DEV v16h loadA_lds(const _Float16* my, int ldh, int kt) {
  const int lane = threadIdx.x & 31;
  const int ar = lane & 15, g = lane >> 4;
  v16h a;
#pragma unroll
  for (int half = 0; half < 2; ++half) {
    const _Float16* p = my + ar * ldh + kt * 32 + half * 16 + g * 8;
#pragma unroll
    for (int j = 0; j < 8; ++j) a[half * 8 + j] = p[j];
  }
  return a;
}

// ---------------------------------------------------------------------------
// Weight pack: W[K][N] f32 -> Wt[N][Kpad] f16 (zero padded)
// ---------------------------------------------------------------------------
__global__ void pack_kernel(const float* __restrict__ W, _Float16* __restrict__ Wt,
                            int K, int N, int Kpad) {
  int i = blockIdx.x * blockDim.x + threadIdx.x;
  if (i >= N * Kpad) return;
  int n = i / Kpad, k = i % Kpad;
  Wt[i] = (k < K) ? (_Float16)W[(size_t)k * N + n] : (_Float16)0.f;
}

__global__ void zero_kernel(float* __restrict__ p, long long n) {
  long long i = (long long)blockIdx.x * blockDim.x + threadIdx.x;
  long long stride = (long long)gridDim.x * blockDim.x;
  for (; i < n; i += stride) p[i] = 0.f;
}

// ---------------------------------------------------------------------------
// enc_n: (65536x32) @ (32x64) + b  -> encN (f32)
// ---------------------------------------------------------------------------
__global__ void enc_n_kernel(const float* __restrict__ X, const _Float16* __restrict__ Wt,
                             const float* __restrict__ b, float* __restrict__ out) {
  const int wave = threadIdx.x >> 5;
  const int row0 = (blockIdx.x * 4 + wave) * 16;
  if (row0 >= kNodes) return;
  const int lane = threadIdx.x & 31, c = lane & 15, g = lane >> 4;
  v16h a;
#pragma unroll
  for (int half = 0; half < 2; ++half) {
    const float* p = X + (size_t)(row0 + c) * 32 + half * 16 + g * 8;
#pragma unroll
    for (int j = 0; j < 8; ++j) a[half * 8 + j] = (_Float16)p[j];
  }
#pragma unroll
  for (int ct = 0; ct < 4; ++ct) {
    v8f acc = wmma16(a, loadB(Wt, 32, ct, 0), vzero());
    float bias = b[ct * 16 + c];
#pragma unroll
    for (int v = 0; v < 8; ++v)
      out[(size_t)(row0 + g * 8 + v) * 64 + ct * 16 + c] = acc[v] + bias;
  }
}

// ---------------------------------------------------------------------------
// enc_e: (131072x16) @ (16x16) + b  (tiny -> plain VALU)
// ---------------------------------------------------------------------------
__global__ void enc_e_kernel(const float* __restrict__ E, const float* __restrict__ W,
                             const float* __restrict__ b, float* __restrict__ out) {
  int e = blockIdx.x * blockDim.x + threadIdx.x;
  if (e >= kEdges) return;
  float x[16];
#pragma unroll
  for (int i = 0; i < 16; ++i) x[i] = E[(size_t)e * 16 + i];
#pragma unroll
  for (int j = 0; j < 16; ++j) {
    float s = b[j];
#pragma unroll
    for (int i = 0; i < 16; ++i) s += x[i] * W[i * 16 + j];
    out[(size_t)e * 16 + j] = s;
  }
}

// ---------------------------------------------------------------------------
// comb MLP: [h(64) | inter(64)] -> 128 relu -> 64
// ---------------------------------------------------------------------------
__global__ void comb_kernel(const float* __restrict__ h, const float* __restrict__ inter,
                            int ldI,
                            const _Float16* __restrict__ W1t, const float* __restrict__ b1,
                            const _Float16* __restrict__ W2t, const float* __restrict__ b2,
                            float* __restrict__ out) {
  __shared__ alignas(16) _Float16 lds[4][2048];
  const int wave = threadIdx.x >> 5;
  const int row0 = (blockIdx.x * 4 + wave) * 16;
  if (row0 >= kNodes) return;
  const int lane = threadIdx.x & 31, c = lane & 15, g = lane >> 4;
  _Float16* my = lds[wave];

  v8f acc[8];
#pragma unroll
  for (int i = 0; i < 8; ++i) acc[i] = vzero();
#pragma unroll
  for (int kt = 0; kt < 4; ++kt) {
    v16h a;
    const int r = row0 + c;
#pragma unroll
    for (int half = 0; half < 2; ++half) {
      const int k0 = kt * 32 + half * 16 + g * 8;
      const float* p = (k0 < 64) ? (h + (size_t)r * 64 + k0)
                                 : (inter + (size_t)r * ldI + (k0 - 64));
#pragma unroll
      for (int j = 0; j < 8; ++j) a[half * 8 + j] = (_Float16)p[j];
    }
#pragma unroll
    for (int ct = 0; ct < 8; ++ct) acc[ct] = wmma16(a, loadB(W1t, 128, ct, kt), acc[ct]);
  }
#pragma unroll
  for (int ct = 0; ct < 8; ++ct) {
    float bias = b1[ct * 16 + c];
#pragma unroll
    for (int v = 0; v < 8; ++v) {
      float x = acc[ct][v] + bias;
      my[(g * 8 + v) * 128 + ct * 16 + c] = (_Float16)fmaxf(x, 0.f);
    }
  }
  v8f a2[4];
#pragma unroll
  for (int i = 0; i < 4; ++i) a2[i] = vzero();
#pragma unroll
  for (int kt = 0; kt < 4; ++kt) {
    v16h a = loadA_lds(my, 128, kt);
#pragma unroll
    for (int ct = 0; ct < 4; ++ct) a2[ct] = wmma16(a, loadB(W2t, 128, ct, kt), a2[ct]);
  }
#pragma unroll
  for (int ct = 0; ct < 4; ++ct) {
    float bias = b2[ct * 16 + c];
#pragma unroll
    for (int v = 0; v < 8; ++v)
      out[(size_t)(row0 + g * 8 + v) * 64 + ct * 16 + c] = a2[ct][v] + bias;
  }
}

// ---------------------------------------------------------------------------
// Edge message MLP: [h[s](64) | h[d](64) | e(16) | pad] K=160 -> 128 relu -> 128
// ---------------------------------------------------------------------------
DEV void mlp_edge(const float* __restrict__ h, const float* __restrict__ encE,
                  const int* __restrict__ sIdx, const int* __restrict__ dIdx,
                  int row0,
                  const _Float16* __restrict__ W1t, const float* __restrict__ b1,
                  const _Float16* __restrict__ W2t, const float* __restrict__ b2,
                  _Float16* my, v8f* out) {
  const int lane = threadIdx.x & 31, c = lane & 15, g = lane >> 4;
  v8f acc[8];
#pragma unroll
  for (int i = 0; i < 8; ++i) acc[i] = vzero();
#pragma unroll
  for (int kt = 0; kt < 5; ++kt) {
    v16h a;
    const int r = row0 + c;
#pragma unroll
    for (int half = 0; half < 2; ++half) {
      const int k0 = kt * 32 + half * 16 + g * 8;
      float tmp[8];
      if (k0 < 64) {
        const float* p = h + (size_t)sIdx[r] * 64 + k0;
#pragma unroll
        for (int j = 0; j < 8; ++j) tmp[j] = p[j];
      } else if (k0 < 128) {
        const float* p = h + (size_t)dIdx[r] * 64 + (k0 - 64);
#pragma unroll
        for (int j = 0; j < 8; ++j) tmp[j] = p[j];
      } else if (k0 < 144) {
        const float* p = encE + (size_t)r * 16 + (k0 - 128);
#pragma unroll
        for (int j = 0; j < 8; ++j) tmp[j] = p[j];
      } else {
#pragma unroll
        for (int j = 0; j < 8; ++j) tmp[j] = 0.f;
      }
#pragma unroll
      for (int j = 0; j < 8; ++j) a[half * 8 + j] = (_Float16)tmp[j];
    }
#pragma unroll
    for (int ct = 0; ct < 8; ++ct) acc[ct] = wmma16(a, loadB(W1t, 160, ct, kt), acc[ct]);
  }
#pragma unroll
  for (int ct = 0; ct < 8; ++ct) {
    float bias = b1[ct * 16 + c];
#pragma unroll
    for (int v = 0; v < 8; ++v) {
      float x = acc[ct][v] + bias;
      my[(g * 8 + v) * 128 + ct * 16 + c] = (_Float16)fmaxf(x, 0.f);
    }
  }
#pragma unroll
  for (int i = 0; i < 8; ++i) out[i] = vzero();
#pragma unroll
  for (int kt = 0; kt < 4; ++kt) {
    v16h a = loadA_lds(my, 128, kt);
#pragma unroll
    for (int ct = 0; ct < 8; ++ct) out[ct] = wmma16(a, loadB(W2t, 128, ct, kt), out[ct]);
  }
#pragma unroll
  for (int ct = 0; ct < 8; ++ct) {
    float bias = b2[ct * 16 + c];
#pragma unroll
    for (int v = 0; v < 8; ++v) out[ct][v] += bias;
  }
}

__global__ void edge_scatter_kernel(
    const float* __restrict__ h, const float* __restrict__ encE,
    const int* __restrict__ fromI, const int* __restrict__ toI,
    const _Float16* mW1t, const float* mb1, const _Float16* mW2t, const float* mb2,
    const _Float16* rW1t, const float* rb1, const _Float16* rW2t, const float* rb2,
    float* __restrict__ agg) {
  __shared__ alignas(16) _Float16 lds[4][2048];
  const int wave = threadIdx.x >> 5;
  const int row0 = (blockIdx.x * 4 + wave) * 16;
  if (row0 >= kEdges) return;
  const int lane = threadIdx.x & 31, c = lane & 15, g = lane >> 4;
  v8f m[8];
  // forward: [src, dst, e] -> scatter to to_idx
  mlp_edge(h, encE, fromI, toI, row0, mW1t, mb1, mW2t, mb2, lds[wave], m);
#pragma unroll
  for (int v = 0; v < 8; ++v) {
    int node = toI[row0 + g * 8 + v];
#pragma unroll
    for (int ct = 0; ct < 8; ++ct)
      atomicAdd(&agg[(size_t)node * 128 + ct * 16 + c], m[ct][v]);
  }
  // backward: [dst, src, e] -> scatter to from_idx
  mlp_edge(h, encE, toI, fromI, row0, rW1t, rb1, rW2t, rb2, lds[wave], m);
#pragma unroll
  for (int v = 0; v < 8; ++v) {
    int node = fromI[row0 + g * 8 + v];
#pragma unroll
    for (int ct = 0; ct < 8; ++ct)
      atomicAdd(&agg[(size_t)node * 128 + ct * 16 + c], m[ct][v]);
  }
}

__global__ void edge_final_kernel(
    const float* __restrict__ h, const float* __restrict__ encE,
    const int* __restrict__ fromI, const int* __restrict__ toI,
    const _Float16* mW1t, const float* mb1, const _Float16* mW2t, const float* mb2,
    const _Float16* rW1t, const float* rb1, const _Float16* rW2t, const float* rb2,
    float* __restrict__ out0, float* __restrict__ out1) {
  __shared__ alignas(16) _Float16 lds[4][2048];
  const int wave = threadIdx.x >> 5;
  const int row0 = (blockIdx.x * 4 + wave) * 16;
  if (row0 >= kEdges) return;
  const int lane = threadIdx.x & 31, c = lane & 15, g = lane >> 4;
  v8f mf[8], mr[8];
  mlp_edge(h, encE, fromI, toI, row0, mW1t, mb1, mW2t, mb2, lds[wave], mf);
  mlp_edge(h, encE, toI, fromI, row0, rW1t, rb1, rW2t, rb2, lds[wave], mr);
#pragma unroll
  for (int v = 0; v < 8; ++v) {
    int e = row0 + g * 8 + v;
    int ge = e >> 6, we = e & 63;
    float* dst = (ge & 1) ? (out1 + ((size_t)(ge >> 1) * 64 + we) * 128)
                          : (out0 + ((size_t)(ge >> 1) * 64 + we) * 128);
#pragma unroll
    for (int ct = 0; ct < 8; ++ct) dst[ct * 16 + c] = mf[ct][v] + mr[ct][v];
  }
}

// ---------------------------------------------------------------------------
// upd MLP + residual: [comb(64) | agg(128)] K=192 -> 128 relu -> 64 ; h=comb+out
// ---------------------------------------------------------------------------
__global__ void upd_kernel(const float* __restrict__ comb, const float* __restrict__ agg,
                           const _Float16* __restrict__ W1t, const float* __restrict__ b1,
                           const _Float16* __restrict__ W2t, const float* __restrict__ b2,
                           float* __restrict__ hOut, float* __restrict__ storeOut, int ldS) {
  __shared__ alignas(16) _Float16 lds[4][2048];
  const int wave = threadIdx.x >> 5;
  const int row0 = (blockIdx.x * 4 + wave) * 16;
  if (row0 >= kNodes) return;
  const int lane = threadIdx.x & 31, c = lane & 15, g = lane >> 4;
  _Float16* my = lds[wave];

  v8f acc[8];
#pragma unroll
  for (int i = 0; i < 8; ++i) acc[i] = vzero();
#pragma unroll
  for (int kt = 0; kt < 6; ++kt) {
    v16h a;
    const int r = row0 + c;
#pragma unroll
    for (int half = 0; half < 2; ++half) {
      const int k0 = kt * 32 + half * 16 + g * 8;
      const float* p = (k0 < 64) ? (comb + (size_t)r * 64 + k0)
                                 : (agg + (size_t)r * 128 + (k0 - 64));
#pragma unroll
      for (int j = 0; j < 8; ++j) a[half * 8 + j] = (_Float16)p[j];
    }
#pragma unroll
    for (int ct = 0; ct < 8; ++ct) acc[ct] = wmma16(a, loadB(W1t, 192, ct, kt), acc[ct]);
  }
#pragma unroll
  for (int ct = 0; ct < 8; ++ct) {
    float bias = b1[ct * 16 + c];
#pragma unroll
    for (int v = 0; v < 8; ++v) {
      float x = acc[ct][v] + bias;
      my[(g * 8 + v) * 128 + ct * 16 + c] = (_Float16)fmaxf(x, 0.f);
    }
  }
  v8f a2[4];
#pragma unroll
  for (int i = 0; i < 4; ++i) a2[i] = vzero();
#pragma unroll
  for (int kt = 0; kt < 4; ++kt) {
    v16h a = loadA_lds(my, 128, kt);
#pragma unroll
    for (int ct = 0; ct < 4; ++ct) a2[ct] = wmma16(a, loadB(W2t, 128, ct, kt), a2[ct]);
  }
#pragma unroll
  for (int ct = 0; ct < 4; ++ct) {
    float bias = b2[ct * 16 + c];
#pragma unroll
    for (int v = 0; v < 8; ++v) {
      int r = row0 + g * 8 + v, col = ct * 16 + c;
      float val = comb[(size_t)r * 64 + col] + a2[ct][v] + bias;
      hOut[(size_t)r * 64 + col] = val;
      storeOut[(size_t)r * ldS + col] = val;
    }
  }
}

// ---------------------------------------------------------------------------
// t MLP: h(64) -> 16 relu -> 16 (K padded to 32 in layer 2)
// ---------------------------------------------------------------------------
__global__ void t_kernel(const float* __restrict__ h,
                         const _Float16* __restrict__ W1t, const float* __restrict__ b1,
                         const _Float16* __restrict__ W2t, const float* __restrict__ b2,
                         float* __restrict__ out) {
  __shared__ alignas(16) _Float16 lds[4][512];
  const int wave = threadIdx.x >> 5;
  const int row0 = (blockIdx.x * 4 + wave) * 16;
  if (row0 >= kNodes) return;
  const int lane = threadIdx.x & 31, c = lane & 15, g = lane >> 4;
  _Float16* my = lds[wave];

  v8f acc = vzero();
#pragma unroll
  for (int kt = 0; kt < 2; ++kt) {
    v16h a;
#pragma unroll
    for (int half = 0; half < 2; ++half) {
      const float* p = h + (size_t)(row0 + c) * 64 + kt * 32 + half * 16 + g * 8;
#pragma unroll
      for (int j = 0; j < 8; ++j) a[half * 8 + j] = (_Float16)p[j];
    }
    acc = wmma16(a, loadB(W1t, 64, 0, kt), acc);
  }
  float bias = b1[c];
#pragma unroll
  for (int v = 0; v < 8; ++v) {
    float x = acc[v] + bias;
    my[(g * 8 + v) * 32 + c] = (_Float16)fmaxf(x, 0.f);
    my[(g * 8 + v) * 32 + 16 + c] = (_Float16)0.f;  // K padding 16..31
  }
  v8f a2 = vzero();
  {
    v16h a = loadA_lds(my, 32, 0);
    a2 = wmma16(a, loadB(W2t, 32, 0, 0), a2);
  }
  float bias2 = b2[c];
#pragma unroll
  for (int v = 0; v < 8; ++v)
    out[(size_t)(row0 + g * 8 + v) * 16 + c] = a2[v] + bias2;
}

// ---------------------------------------------------------------------------
// Sinkhorn: logits = tq . tc / TEMP, 20 iters row/col logsumexp, exp -> plan
// One block per graph pair: (32,32) threads = 32 waves, one wave per q-row.
// ---------------------------------------------------------------------------
__global__ void plan_kernel(const float* __restrict__ tAll, float* __restrict__ plan) {
  const int gp = blockIdx.x;
  const int q = threadIdx.y, c = threadIdx.x;
  __shared__ float tcs[32][16];
  __shared__ float mat[32][33];
  __shared__ float colv[32];
  int tid = q * 32 + c;
  const float* tcBase = tAll + ((size_t)(2 * gp + 1) * 32) * 16;
  if (tid < 512) tcs[tid >> 4][tid & 15] = tcBase[tid];
  __syncthreads();
  const float* tqRow = tAll + ((size_t)(2 * gp) * 32 + q) * 16;
  float dot = 0.f;
#pragma unroll
  for (int d = 0; d < 16; ++d) dot += tqRow[d] * tcs[c][d];
  float la = dot * kInvTemp;

  for (int it = 0; it < 20; ++it) {
    // logsumexp over c (within wave = row q)
    float m = la;
#pragma unroll
    for (int off = 16; off >= 1; off >>= 1) m = fmaxf(m, __shfl_xor(m, off, 32));
    float s = __expf(la - m);
#pragma unroll
    for (int off = 16; off >= 1; off >>= 1) s += __shfl_xor(s, off, 32);
    la -= m + __logf(s);
    // logsumexp over q (LDS transpose; wave y reduces column y)
    mat[q][c] = la;
    __syncthreads();
    float v2 = mat[c][q];
    float m2 = v2;
#pragma unroll
    for (int off = 16; off >= 1; off >>= 1) m2 = fmaxf(m2, __shfl_xor(m2, off, 32));
    float s2 = __expf(v2 - m2);
#pragma unroll
    for (int off = 16; off >= 1; off >>= 1) s2 += __shfl_xor(s2, off, 32);
    if (c == 0) colv[q] = m2 + __logf(s2);
    __syncthreads();
    la -= colv[c];
  }
  plan[((size_t)gp * 32 + q) * 32 + c] = __expf(la);
}

// ---------------------------------------------------------------------------
// Interaction: q_from_c = plan @ c, c_from_q = plan^T @ q; keep cols 64..255,
// zero cols 0..63. Reads S1, writes S0.
// ---------------------------------------------------------------------------
__global__ void interaction_kernel(const float* __restrict__ S, const float* __restrict__ plan,
                                   float* __restrict__ O) {
  const int g = blockIdx.x;
  __shared__ float pl[1024];
  int tid = threadIdx.x;
  for (int i = tid; i < 1024; i += 256) pl[i] = plan[(size_t)g * 1024 + i];
  __syncthreads();
  const float* qm = S + (size_t)(2 * g) * 32 * 256;
  const float* cm = S + (size_t)(2 * g + 1) * 32 * 256;
  float* oq = O + (size_t)(2 * g) * 32 * 256;
  float* oc = O + (size_t)(2 * g + 1) * 32 * 256;
  for (int i = tid; i < 4096; i += 256) {
    int r = i >> 6, col = i & 63;
    if (r < 32) oq[r * 256 + col] = 0.f;
    else        oc[(r - 32) * 256 + col] = 0.f;
  }
  for (int i = tid; i < 32 * 192; i += 256) {
    int r = i / 192, d = 64 + i % 192;
    float aq = 0.f, ac = 0.f;
#pragma unroll 8
    for (int k = 0; k < 32; ++k) {
      aq += pl[r * 32 + k] * cm[k * 256 + d];
      ac += pl[k * 32 + r] * qm[k * 256 + d];
    }
    oq[r * 256 + d] = aq;
    oc[r * 256 + d] = ac;
  }
}

// ---------------------------------------------------------------------------
// Host launch
// ---------------------------------------------------------------------------
extern "C" void kernel_launch(void* const* d_in, const int* in_sizes, int n_in,
                              void* d_out, int out_size, void* d_ws, size_t ws_size,
                              hipStream_t stream) {
  (void)in_sizes; (void)n_in; (void)out_size; (void)ws_size;

  const float* nodeF   = (const float*)d_in[0];
  const float* edgeF   = (const float*)d_in[1];
  const float* enc_n_W = (const float*)d_in[2];
  const float* enc_n_b = (const float*)d_in[3];
  const float* enc_e_W = (const float*)d_in[4];
  const float* enc_e_b = (const float*)d_in[5];
  const float* msg_W1  = (const float*)d_in[6];
  const float* msg_b1  = (const float*)d_in[7];
  const float* msg_W2  = (const float*)d_in[8];
  const float* msg_b2  = (const float*)d_in[9];
  const float* rmsg_W1 = (const float*)d_in[10];
  const float* rmsg_b1 = (const float*)d_in[11];
  const float* rmsg_W2 = (const float*)d_in[12];
  const float* rmsg_b2 = (const float*)d_in[13];
  const float* upd_W1  = (const float*)d_in[14];
  const float* upd_b1  = (const float*)d_in[15];
  const float* upd_W2  = (const float*)d_in[16];
  const float* upd_b2  = (const float*)d_in[17];
  const float* comb_W1 = (const float*)d_in[18];
  const float* comb_b1 = (const float*)d_in[19];
  const float* comb_W2 = (const float*)d_in[20];
  const float* comb_b2 = (const float*)d_in[21];
  const float* t1_W    = (const float*)d_in[22];
  const float* t1_b    = (const float*)d_in[23];
  const float* t2_W    = (const float*)d_in[24];
  const float* t2_b    = (const float*)d_in[25];
  const int* fromI     = (const int*)d_in[26];
  const int* toI       = (const int*)d_in[27];

  char* w = (char*)d_ws;
  auto alloc = [&](size_t bytes) -> void* {
    void* p = (void*)w;
    w += (bytes + 255) & ~(size_t)255;
    return p;
  };
  float* encN   = (float*)alloc((size_t)kNodes * 64 * 4);
  float* encE   = (float*)alloc((size_t)kEdges * 16 * 4);
  float* S0     = (float*)alloc((size_t)kNodes * 256 * 4);
  float* S1     = (float*)alloc((size_t)kNodes * 256 * 4);
  float* hBuf   = (float*)alloc((size_t)kNodes * 64 * 4);
  float* combB  = (float*)alloc((size_t)kNodes * 64 * 4);
  float* agg    = (float*)alloc((size_t)kNodes * 128 * 4);
  float* tAll   = (float*)alloc((size_t)kNodes * 16 * 4);
  _Float16* encnWt  = (_Float16*)alloc((size_t)64 * 32 * 2);
  _Float16* combW1t = (_Float16*)alloc((size_t)128 * 128 * 2);
  _Float16* combW2t = (_Float16*)alloc((size_t)64 * 128 * 2);
  _Float16* msgW1t  = (_Float16*)alloc((size_t)128 * 160 * 2);
  _Float16* msgW2t  = (_Float16*)alloc((size_t)128 * 128 * 2);
  _Float16* rmsgW1t = (_Float16*)alloc((size_t)128 * 160 * 2);
  _Float16* rmsgW2t = (_Float16*)alloc((size_t)128 * 128 * 2);
  _Float16* updW1t  = (_Float16*)alloc((size_t)128 * 192 * 2);
  _Float16* updW2t  = (_Float16*)alloc((size_t)64 * 128 * 2);
  _Float16* t1t     = (_Float16*)alloc((size_t)16 * 64 * 2);
  _Float16* t2t     = (_Float16*)alloc((size_t)16 * 32 * 2);

  auto pack = [&](const float* W, _Float16* Wt, int K, int N, int Kpad) {
    int n = N * Kpad;
    pack_kernel<<<(n + 255) / 256, 256, 0, stream>>>(W, Wt, K, N, Kpad);
  };
  pack(enc_n_W, encnWt, 32, 64, 32);
  pack(comb_W1, combW1t, 128, 128, 128);
  pack(comb_W2, combW2t, 128, 64, 128);
  pack(msg_W1, msgW1t, 144, 128, 160);
  pack(msg_W2, msgW2t, 128, 128, 128);
  pack(rmsg_W1, rmsgW1t, 144, 128, 160);
  pack(rmsg_W2, rmsgW2t, 128, 128, 128);
  pack(upd_W1, updW1t, 192, 128, 192);
  pack(upd_W2, updW2t, 128, 64, 128);
  pack(t1_W, t1t, 64, 16, 64);
  pack(t2_W, t2t, 16, 16, 32);

  zero_kernel<<<2048, 256, 0, stream>>>(S0, (long long)kNodes * 256);
  enc_n_kernel<<<kNodes / 64, 128, 0, stream>>>(nodeF, encnWt, enc_n_b, encN);
  enc_e_kernel<<<kEdges / 256, 256, 0, stream>>>(edgeF, enc_e_W, enc_e_b, encE);

  float* out0    = (float*)d_out;
  float* out1    = out0 + (size_t)kG * 64 * 128;
  float* planOut = out1 + (size_t)kG * 64 * 128;

  for (int t = 0; t < 2; ++t) {
    const float* hIn = encN;
    for (int k = 1; k <= 3; ++k) {
      comb_kernel<<<kNodes / 64, 128, 0, stream>>>(
          hIn, S0 + (size_t)(k - 1) * 64, 256,
          combW1t, comb_b1, combW2t, comb_b2, combB);
      zero_kernel<<<2048, 256, 0, stream>>>(agg, (long long)kNodes * 128);
      edge_scatter_kernel<<<kEdges / 64, 128, 0, stream>>>(
          combB, encE, fromI, toI,
          msgW1t, msg_b1, msgW2t, msg_b2,
          rmsgW1t, rmsg_b1, rmsgW2t, rmsg_b2, agg);
      upd_kernel<<<kNodes / 64, 128, 0, stream>>>(
          combB, agg, updW1t, upd_b1, updW2t, upd_b2,
          hBuf, S1 + (size_t)k * 64, 256);
      hIn = hBuf;
    }
    t_kernel<<<kNodes / 64, 128, 0, stream>>>(hBuf, t1t, t1_b, t2t, t2_b, tAll);
    plan_kernel<<<kG, dim3(32, 32), 0, stream>>>(tAll, planOut);
    interaction_kernel<<<kG, 256, 0, stream>>>(S1, planOut, S0);
  }
  // final: comb([h, store[:, -64:]]) then forward+backward message to outputs
  comb_kernel<<<kNodes / 64, 128, 0, stream>>>(
      hBuf, S0 + 192, 256, combW1t, comb_b1, combW2t, comb_b2, combB);
  edge_final_kernel<<<kEdges / 64, 128, 0, stream>>>(
      combB, encE, fromI, toI,
      msgW1t, msg_b1, msgW2t, msg_b2,
      rmsgW1t, rmsg_b1, rmsgW2t, rmsg_b2, out0, out1);
}
